// DynamicCombiner_55259049230428
// MI455X (gfx1250) — compile-verified, matrix operand order checked by reading, not run on hardware
//
#include <hip/hip_runtime.h>

// Problem constants (B=8, S=256, D=1024, V=32000, K=32 -> N=2048)
#define NROWS 2048
#define DDIM  1024
#define FDIM  2048
#define VDIM  32000
#define V4DIM 8000
#define KNN   32

typedef __attribute__((ext_vector_type(16))) __bf16 v16bf;
typedef __attribute__((ext_vector_type(8)))  __bf16 v8bf;
typedef __attribute__((ext_vector_type(4)))  __bf16 v4bf;
typedef __attribute__((ext_vector_type(8)))  float  v8f;
typedef __attribute__((ext_vector_type(4)))  float  v4f;

// ---------------------------------------------------------------------------
// Kernel 1: convert W1 (mw_w1, [1024 x 2048] f32) to bf16 for WMMA
// ---------------------------------------------------------------------------
__global__ void k_conv_w1(const float* __restrict__ w1, __bf16* __restrict__ w1b) {
  int i = blockIdx.x * blockDim.x + threadIdx.x;   // [0, 2M/4)
  v4f x = ((const v4f*)w1)[i];
  v4bf o;
  o.x = (__bf16)x.x; o.y = (__bf16)x.y; o.z = (__bf16)x.z; o.w = (__bf16)x.w;
  ((v4bf*)w1b)[i] = o;
}

// ---------------------------------------------------------------------------
// Kernel 2: per-row prep. One 256-thread workgroup per row n.
//   ctx = mean_k searched_hidden[n,k,:]   (NT loads: stream-once, 268MB)
//   feat = [h, ctx]  -> bf16 workspace
//   bandwidth = exp(feat . bw_w + bw_b)
//   knn_w = softmax(-dist/bandwidth) over K=32 (first wave)
//   mixdot[n] = 0 (accumulator for GEMM atomics)
// ---------------------------------------------------------------------------
__global__ void k_rowprep(const float* __restrict__ hidden,
                          const float* __restrict__ srch,
                          const float* __restrict__ dist,
                          const float* __restrict__ bw_w,
                          const float* __restrict__ bw_b,
                          __bf16* __restrict__ featb,
                          float* __restrict__ knn_w,
                          float* __restrict__ mixdot) {
  __shared__ float red[256];
  const int n = blockIdx.x;
  const int t = threadIdx.x;
  const int c0 = t * 4;

  // mean over K of searched_hidden[n, :, c0..c0+3]
  const v4f* s4 = (const v4f*)(srch + (size_t)n * KNN * DDIM);
  v4f acc = {0.f, 0.f, 0.f, 0.f};
#pragma unroll 4
  for (int k = 0; k < KNN; ++k) {
    v4f x = __builtin_nontemporal_load(&s4[(k * DDIM + c0) >> 2]);
    acc.x += x.x; acc.y += x.y; acc.z += x.z; acc.w += x.w;
  }
  const float inv = 1.0f / (float)KNN;
  acc.x *= inv; acc.y *= inv; acc.z *= inv; acc.w *= inv;

  v4f h = ((const v4f*)(hidden + (size_t)n * DDIM))[t];

  // store feat as bf16: cols [0,1024) = h, [1024,2048) = ctx
  __bf16* fr = featb + (size_t)n * FDIM;
  v4bf hb; hb.x = (__bf16)h.x; hb.y = (__bf16)h.y; hb.z = (__bf16)h.z; hb.w = (__bf16)h.w;
  ((v4bf*)fr)[t] = hb;
  v4bf cb; cb.x = (__bf16)acc.x; cb.y = (__bf16)acc.y; cb.z = (__bf16)acc.z; cb.w = (__bf16)acc.w;
  ((v4bf*)(fr + DDIM))[t] = cb;

  // bandwidth dot partial
  float d = h.x * bw_w[c0]   + h.y * bw_w[c0+1] + h.z * bw_w[c0+2] + h.w * bw_w[c0+3]
          + acc.x * bw_w[DDIM+c0]   + acc.y * bw_w[DDIM+c0+1]
          + acc.z * bw_w[DDIM+c0+2] + acc.w * bw_w[DDIM+c0+3];
  red[t] = d;
  __syncthreads();
  for (int o = 128; o > 0; o >>= 1) {
    if (t < o) red[t] += red[t + o];
    __syncthreads();
  }
  const float bw = __expf(red[0] + bw_b[0]);

  if (t < KNN) {  // first wave: softmax over 32 distances
    float v = -dist[(size_t)n * KNN + t] / bw;
    float m = v;
#pragma unroll
    for (int s = 16; s > 0; s >>= 1) m = fmaxf(m, __shfl_xor(m, s, 32));
    float e = __expf(v - m);
    float sum = e;
#pragma unroll
    for (int s = 16; s > 0; s >>= 1) sum += __shfl_xor(sum, s, 32);
    knn_w[(size_t)n * KNN + t] = e / sum;
  }
  if (t == 0) mixdot[n] = 0.0f;
}

// ---------------------------------------------------------------------------
// Kernel 3: WMMA bf16 GEMM fused with relu/w2 epilogue.
//   mixdot[n] += sum_j relu(feat[n].W1[j] + b1[j]) * w2[j]
// Each wave computes a 32(M)x32(J) output tile: 2 A-frags x 2 B-frags ->
// 4 independent WMMAs per k-step (8 b128 loads), halving operand bytes/WMMA
// and improving XDL/VALU co-execution. 64x32 tiles = 2048 wave-tasks,
// 8 waves/block -> 256 blocks. K cannot be split (ReLU before the w2 dot).
// A/B frag (bf16 16x32 / 32x16) striping per ISA 7.12.2: lane l ->
// row/col = l&15, hi = l>>4; two 8-elem contiguous chunks at k0+hi*8 and
// k0+16+hi*8.
// ---------------------------------------------------------------------------
__global__ void k_mixgemm(const __bf16* __restrict__ featb,
                          const __bf16* __restrict__ w1b,
                          const float* __restrict__ b1,
                          const float* __restrict__ w2,
                          float* __restrict__ mixdot) {
  const int wave = threadIdx.x >> 5;
  const int lane = threadIdx.x & 31;
  const int tile = blockIdx.x * 8 + wave;   // 0..2047
  const int mt   = tile >> 5;               // 0..63  (32-row M block)
  const int jt   = tile & 31;               // 0..31  (32-col J block)
  const int mBase = mt * 32;
  const int jBase = jt * 32;
  const int m  = lane & 15;
  const int hi = lane >> 4;

  const __bf16* ra0 = featb + (size_t)(mBase + m) * FDIM;
  const __bf16* ra1 = ra0 + (size_t)16 * FDIM;
  const __bf16* rb0 = w1b + (size_t)(jBase + m) * FDIM;
  const __bf16* rb1 = rb0 + (size_t)16 * FDIM;

  v8f acc00 = {0.f,0.f,0.f,0.f,0.f,0.f,0.f,0.f};
  v8f acc01 = acc00, acc10 = acc00, acc11 = acc00;

  for (int k0 = 0; k0 < FDIM; k0 += 32) {
    const int kA = k0 + hi * 8;
    v8bf a0lo = *(const v8bf*)(ra0 + kA);
    v8bf a0hi = *(const v8bf*)(ra0 + kA + 16);
    v8bf a1lo = *(const v8bf*)(ra1 + kA);
    v8bf a1hi = *(const v8bf*)(ra1 + kA + 16);
    v8bf b0lo = *(const v8bf*)(rb0 + kA);
    v8bf b0hi = *(const v8bf*)(rb0 + kA + 16);
    v8bf b1lo = *(const v8bf*)(rb1 + kA);
    v8bf b1hi = *(const v8bf*)(rb1 + kA + 16);
    v16bf a0 = __builtin_shufflevector(a0lo, a0hi, 0,1,2,3,4,5,6,7,8,9,10,11,12,13,14,15);
    v16bf a1 = __builtin_shufflevector(a1lo, a1hi, 0,1,2,3,4,5,6,7,8,9,10,11,12,13,14,15);
    v16bf b0 = __builtin_shufflevector(b0lo, b0hi, 0,1,2,3,4,5,6,7,8,9,10,11,12,13,14,15);
    v16bf b1 = __builtin_shufflevector(b1lo, b1hi, 0,1,2,3,4,5,6,7,8,9,10,11,12,13,14,15);
    acc00 = __builtin_amdgcn_wmma_f32_16x16x32_bf16(false, a0, false, b0, (short)0, acc00, false, false);
    acc01 = __builtin_amdgcn_wmma_f32_16x16x32_bf16(false, a0, false, b1, (short)0, acc01, false, false);
    acc10 = __builtin_amdgcn_wmma_f32_16x16x32_bf16(false, a1, false, b0, (short)0, acc10, false, false);
    acc11 = __builtin_amdgcn_wmma_f32_16x16x32_bf16(false, a1, false, b1, (short)0, acc11, false, false);
  }

  // C/D layout: vgpr r, lanes0-15 -> row r, lanes16-31 -> row 8+r;
  // column = (lane&15). Fuse bias -> relu -> *w2, accumulate per-row.
  float part[2][8];
#pragma unroll
  for (int mi = 0; mi < 2; ++mi)
#pragma unroll
    for (int r = 0; r < 8; ++r) part[mi][r] = 0.0f;

  {
    const int j0 = jBase + m;
    const float bj0 = b1[j0], wj0 = w2[j0];
    const int j1 = jBase + 16 + m;
    const float bj1 = b1[j1], wj1 = w2[j1];
#pragma unroll
    for (int r = 0; r < 8; ++r) {
      part[0][r] += fmaxf(acc00[r] + bj0, 0.0f) * wj0;
      part[0][r] += fmaxf(acc01[r] + bj1, 0.0f) * wj1;
      part[1][r] += fmaxf(acc10[r] + bj0, 0.0f) * wj0;
      part[1][r] += fmaxf(acc11[r] + bj1, 0.0f) * wj1;
    }
  }

  // reduce over the 16 lanes of each half (xor masks < 16 stay in-half)
#pragma unroll
  for (int mi = 0; mi < 2; ++mi) {
#pragma unroll
    for (int r = 0; r < 8; ++r) {
      float v = part[mi][r];
      v += __shfl_xor(v, 1, 32);
      v += __shfl_xor(v, 2, 32);
      v += __shfl_xor(v, 4, 32);
      v += __shfl_xor(v, 8, 32);
      if (m == 0) atomicAdd(&mixdot[mBase + mi * 16 + r + 8 * hi], v);
    }
  }
}

// ---------------------------------------------------------------------------
// Kernel 4: per-row softmax over V + mixed log output (base values).
// One 256-thread workgroup per row. Pass1: online max/sum (branchy form:
// 1 exp on the common no-new-max path). Pass2: write log((1-mix)*p + 1e-10)
// with NT stores. Stashes rowmax/rowsum/mixing for the fixup kernel.
// ---------------------------------------------------------------------------
__global__ void k_softmax(const float* __restrict__ logits,
                          const float* __restrict__ mixdot,
                          const float* __restrict__ mw_b2,
                          float* __restrict__ out,
                          float* __restrict__ mixing,
                          float* __restrict__ rowmax,
                          float* __restrict__ rowsum) {
  __shared__ float sm[256];
  __shared__ float ss[256];
  __shared__ float bc[3];  // M, S, mix
  const int n = blockIdx.x;
  const int t = threadIdx.x;
  const size_t base4 = (size_t)n * V4DIM;
  const v4f* lg4 = (const v4f*)logits + base4;

  float m = -INFINITY, s = 0.0f;
  for (int i = t; i < V4DIM; i += 256) {
    v4f x = lg4[i];
#pragma unroll
    for (int c = 0; c < 4; ++c) {
      float xv = (c == 0) ? x.x : (c == 1) ? x.y : (c == 2) ? x.z : x.w;
      if (xv > m) { s = s * __expf(m - xv) + 1.0f; m = xv; }
      else        { s += __expf(xv - m); }
    }
  }
  sm[t] = m; ss[t] = s;
  __syncthreads();
  for (int o = 128; o > 0; o >>= 1) {
    if (t < o) {
      float m1 = sm[t], s1 = ss[t], m2 = sm[t + o], s2 = ss[t + o];
      float M = fmaxf(m1, m2);
      sm[t] = M;
      ss[t] = s1 * __expf(m1 - M) + s2 * __expf(m2 - M);
    }
    __syncthreads();
  }
  if (t == 0) {
    float z = mixdot[n] + mw_b2[0];
    float mix = 1.0f / (1.0f + __expf(-z));
    bc[0] = sm[0]; bc[1] = ss[0]; bc[2] = mix;
    rowmax[n] = sm[0]; rowsum[n] = ss[0]; mixing[n] = mix;
  }
  __syncthreads();
  const float M = bc[0];
  const float invS = 1.0f / bc[1];
  const float om = 1.0f - bc[2];

  v4f* out4 = (v4f*)out + base4;
  for (int i = t; i < V4DIM; i += 256) {
    v4f x = lg4[i];   // second read: expected L2 hit
    v4f o;
    o.x = __logf(om * __expf(x.x - M) * invS + 1e-10f);
    o.y = __logf(om * __expf(x.y - M) * invS + 1e-10f);
    o.z = __logf(om * __expf(x.z - M) * invS + 1e-10f);
    o.w = __logf(om * __expf(x.w - M) * invS + 1e-10f);
    __builtin_nontemporal_store(o, &out4[i]);
  }
}

// ---------------------------------------------------------------------------
// Kernel 5: scatter fixup. One wave per row; lane k owns (idx_k, w_k).
// Duplicate indices combined via shuffles; the first-occurrence lane
// overwrites out[n, idx] = log((1-mix)*p + mix*e + 1e-10).
// Runs after k_softmax (stream order -> stores visible).
// ---------------------------------------------------------------------------
__global__ void k_fixup(const float* __restrict__ logits,
                        const int* __restrict__ token_indices,
                        const float* __restrict__ knn_w,
                        const float* __restrict__ mixing,
                        const float* __restrict__ rowmax,
                        const float* __restrict__ rowsum,
                        float* __restrict__ out) {
  const int n = blockIdx.x * 8 + (threadIdx.x >> 5);
  const int lane = threadIdx.x & 31;
  const int idx = token_indices[(size_t)n * KNN + lane];
  const float w = knn_w[(size_t)n * KNN + lane];

  float tot = 0.0f;
  bool first = true;
#pragma unroll
  for (int j = 0; j < KNN; ++j) {
    int oi   = __shfl(idx, j, 32);
    float ow = __shfl(w, j, 32);
    if (oi == idx) {
      tot += ow;
      if (j < lane) first = false;
    }
  }
  if (first) {
    const float M = rowmax[n];
    const float S = rowsum[n];
    const float mix = mixing[n];
    const size_t pos = (size_t)n * VDIM + (size_t)idx;
    float p = __expf(logits[pos] - M) / S;
    out[pos] = __logf((1.0f - mix) * p + mix * tot + 1e-10f);
  }
}

// ---------------------------------------------------------------------------
// Host launcher
// ---------------------------------------------------------------------------
extern "C" void kernel_launch(void* const* d_in, const int* in_sizes, int n_in,
                              void* d_out, int out_size, void* d_ws, size_t ws_size,
                              hipStream_t stream) {
  const float* hidden   = (const float*)d_in[0];
  const float* logits   = (const float*)d_in[1];
  const float* dist     = (const float*)d_in[2];
  const int*   tok_idx  = (const int*)d_in[3];
  const float* srch     = (const float*)d_in[4];
  const float* bw_w     = (const float*)d_in[5];
  const float* bw_b     = (const float*)d_in[6];
  const float* mw_w1    = (const float*)d_in[7];
  const float* mw_b1    = (const float*)d_in[8];
  const float* mw_w2    = (const float*)d_in[9];
  const float* mw_b2    = (const float*)d_in[10];
  float* out = (float*)d_out;

  char* ws = (char*)d_ws;
  __bf16* featb  = (__bf16*)(ws);                            // 2048*2048*2 = 8,388,608
  __bf16* w1b    = (__bf16*)(ws + 8388608);                  // 1024*2048*2 = 4,194,304
  float*  knnw   = (float*)(ws + 8388608 + 4194304);         // 65536*4     =   262,144
  float*  mixdot = (float*)(ws + 12845056);                  // 2048*4
  float*  mixw   = (float*)(ws + 12853248);                  // 2048*4
  float*  rmax   = (float*)(ws + 12861440);                  // 2048*4
  float*  rsum   = (float*)(ws + 12869632);                  // 2048*4

  // 1) W1 -> bf16 (2M elems, 4/thread via v4)
  k_conv_w1<<<2048, 256, 0, stream>>>(mw_w1, w1b);
  // 2) per-row prep (ctx mean, feat bf16, bandwidth, knn softmax, zero mixdot)
  k_rowprep<<<NROWS, 256, 0, stream>>>(hidden, srch, dist, bw_w, bw_b,
                                       featb, knnw, mixdot);
  // 3) WMMA GEMM (32x32 tile per wave) fused mixing-dot
  k_mixgemm<<<256, 256, 0, stream>>>(featb, w1b, mw_b1, mw_w2, mixdot);
  // 4) per-row softmax + base output
  k_softmax<<<NROWS, 256, 0, stream>>>(logits, mixdot, mw_b2, out, mixw, rmax, rsum);
  // 5) K-sparse scatter fixup
  k_fixup<<<NROWS / 8, 256, 0, stream>>>(logits, tok_idx, knnw, mixw, rmax, rsum, out);
}
